// NCAWithHyperNetwork_70712341561436
// MI455X (gfx1250) — matrix operand order repaired and checked by reading, not compiled, not run on hardware
//
#include <hip/hip_runtime.h>
#include <hip/hip_bf16.h>

typedef __attribute__((ext_vector_type(16))) __bf16 v16bf;
typedef __attribute__((ext_vector_type(8)))  float  v8f;
typedef __attribute__((ext_vector_type(4)))  unsigned int ui4;
typedef __attribute__((ext_vector_type(8)))  int i8v;
typedef __attribute__((ext_vector_type(4)))  int i4v;

#define CCH   64
#define HH    64
#define WWID  64
#define BATCH 32
#define DDIM  128
#define HID   256
#define TOT1  36864          // 64*64*9
#define NSTEPS 32

#if defined(__has_builtin)
#if __has_builtin(__builtin_amdgcn_tensor_load_to_lds) && __has_builtin(__builtin_amdgcn_s_wait_tensorcnt)
#define HAVE_TDM 1
#endif
#endif

__device__ __forceinline__ unsigned short f32_to_bf16(float f) {
    unsigned int u = __float_as_uint(f);
    u += 0x7FFFu + ((u >> 16) & 1u);       // round-to-nearest-even
    return (unsigned short)(u >> 16);
}

// ---------------- cond gather: cond[b,d] = emb[label[b], d] ----------------
__global__ void cond_kernel(const int* __restrict__ labels,
                            const float* __restrict__ emb,
                            float* __restrict__ cond) {
    int i = blockIdx.x * blockDim.x + threadIdx.x;
    if (i >= BATCH * DDIM) return;
    int b = i >> 7, d = i & 127;
    cond[i] = emb[labels[b] * DDIM + d];
}

// ---------------- generic FC: out[b,n] = (relu)(in[b,:]@w[:,n] + bias[n]) ----------------
__global__ void fc_kernel(const float* __restrict__ in, const float* __restrict__ w,
                          const float* __restrict__ bias, float* __restrict__ out,
                          int K, int N, int relu) {
    long long i = (long long)blockIdx.x * blockDim.x + threadIdx.x;
    long long total = (long long)BATCH * N;
    if (i >= total) return;
    int b = (int)(i / N), n = (int)(i % N);
    float sum = bias[n];
    const float* ip = in + (long long)b * K;
    for (int k = 0; k < K; ++k) sum = fmaf(ip[k], w[(long long)k * N + n], sum);
    out[i] = relu ? fmaxf(sum, 0.0f) : sum;
}

// ---------------- pack hypernet fp32 weights -> bf16 WMMA A-fragment layout ----------------
// frag elem index u = (((m*9 + t)*2 + kc)*32 + lane)*16 + e
// A 16x32 bf16 order used here (contiguous k, B matches): lanes 0-15: M=lane, e0..15 -> K0..15
//                      lanes 16-31: M=lane-16, e0..15 -> K16..31
__global__ void pack_kernel(const float* __restrict__ wraw,  // [B, 2*TOT1]
                            unsigned short* __restrict__ wpack) { // [2][B][TOT1]
    long long i = (long long)blockIdx.x * blockDim.x + threadIdx.x;
    if (i >= (long long)2 * BATCH * TOT1) return;
    int u  = (int)(i % TOT1);
    int sc = (int)(i / TOT1);
    int b  = sc & 31;
    int cv = sc >> 5;
    int e    = u & 15;
    int lane = (u >> 4) & 31;
    int kc   = (u >> 9) & 1;
    int tm   = u >> 10;           // m*9 + t
    int t    = tm % 9;
    int m    = tm / 9;
    int klocal = (lane < 16) ? e
                             : e + 16;
    int ci = kc * 32 + klocal;
    int co = m * 16 + (lane & 15);
    float v = wraw[(long long)b * (2 * TOT1) + (long long)cv * TOT1 + (co * CCH + ci) * 9 + t];
    wpack[(long long)(cv * BATCH + b) * TOT1 + u] = f32_to_bf16(v);
}

// ---------------- WMMA conv step kernel ----------------
// mode 0: gout = relu(g*conv(gin) + be)                       (u1)
// mode 1: u = relu(g*conv(gin) + be); gout = (1-s)*gold + s*u; rgb = sigmoid(gout[:3])
// grid: 1024 blocks = 32 samples * 32 row-pair tiles; 256 threads = 8 wave32
// wave tiling: 2 Cout tiles x 2 pixel tiles (square -> min LDS traffic per WMMA)
__global__ __launch_bounds__(256)
void conv_kernel(const float* __restrict__ gin,
                 const float* __restrict__ gold,
                 float* __restrict__ gout,
                 const unsigned short* __restrict__ wpack,  // this conv's [B][TOT1] frags
                 const float* __restrict__ film,            // [B][128] gamma|beta
                 const float* __restrict__ blend_p,
                 float* __restrict__ traj,                  // mode1: rgb for this step
                 float* __restrict__ final_rgb,             // mode1 step31, else null
                 int mode)
{
    extern __shared__ unsigned char smem[];
    unsigned short* aw  = (unsigned short*)smem;                 // 36864 bf16 = 72KB
    unsigned short* xin = (unsigned short*)(smem + 73728);       // 4*64*64 bf16 = 32KB

    const int tid  = threadIdx.x;
    const int blk  = blockIdx.x;
    const int b    = blk >> 5;
    const int tile = blk & 31;
    const int y0   = tile * 2;

    // ---- stage this sample's pre-packed A fragments into LDS (72KB) ----
#ifdef HAVE_TDM
    if (tid < 32) {   // wave 0 issues one TDM DMA: 9216 x 8B elements, global -> LDS base
        unsigned long long ga = (unsigned long long)(const void*)(wpack + (long long)b * TOT1);
        ui4 g0;
        g0.x = 1u;                                    // count=1, user descriptor
        g0.y = 0u;                                    // lds_addr = dynamic-LDS base
        g0.z = (unsigned)(ga & 0xFFFFFFFFull);        // global_addr[31:0]
        g0.w = (unsigned)((ga >> 32) & 0x01FFFFFFull) | (2u << 30);  // addr[56:32] | type=2
        i8v g1;
        g1[0] = 0x30000;                  // workgroup_mask=0, data_size=3 (8B)
        g1[1] = (int)(9216u << 16);       // tensor_dim0[15:0]=9216 (<<16), abar_addr=0
        g1[2] = 0x10000;                  // tensor_dim0[31:16]=0, tensor_dim1[15:0]=1
        g1[3] = (int)(9216u << 16);       // tensor_dim1[31:16]=0, tile_dim0=9216
        g1[4] = 1;                        // tile_dim1=1, tile_dim2=0
        g1[5] = 9216;                     // tensor_dim0_stride[31:0]
        g1[6] = 0;
        g1[7] = 0;
        i4v gz = {0, 0, 0, 0};
#if __clang_major__ >= 23
        i8v gz8 = {0, 0, 0, 0, 0, 0, 0, 0};
        __builtin_amdgcn_tensor_load_to_lds(g0, g1, gz, gz, gz8, 0);
#else
        __builtin_amdgcn_tensor_load_to_lds(g0, g1, gz, gz, 0);
#endif
    }
#else
    {
        const uint4* src = (const uint4*)(wpack + (long long)b * TOT1);
        uint4* dst = (uint4*)aw;
        #pragma unroll
        for (int i = 0; i < 18; ++i)                  // 18*256 = 4608 uint4 = 72KB
            dst[tid + 256 * i] = src[tid + 256 * i];
    }
#endif

    // ---- stage input rows y0-1..y0+2 (circular) as bf16, layout [(r*64+x)*64 + ci] ----
    for (int i = tid; i < 4 * 64 * 64; i += 256) {
        int x  = i & 63;
        int ci = (i >> 6) & 63;
        int r  = i >> 12;
        int y  = (y0 - 1 + r) & 63;
        float v = gin[(((b * CCH) + ci) * HH + y) * WWID + x];
        xin[((r * 64 + x) * 64) + ci] = f32_to_bf16(v);
    }
#ifdef HAVE_TDM
    __builtin_amdgcn_s_wait_tensorcnt(0);   // no-op for waves with TENSORcnt==0
#endif
    __syncthreads();

    const int wv   = tid >> 5;
    const int lane = tid & 31;
    const int mp   = wv & 1;               // Cout tile pair: m = 2*mp + {0,1}
    const int np   = wv >> 1;              // pixel tile pair (0..3)
    const int py   = np >> 1;              // output row within the 2-row tile
    const int x0   = (np & 1) * 32 + (lane & 15);   // pixel of N-tile 0
    const int hi   = lane >> 4;

    // precompute B-fragment base offsets (elements) for the 3 circular x shifts
    unsigned bb0[3], bb1[3];
    #pragma unroll
    for (int dx = 0; dx < 3; ++dx) {
        bb0[dx] = (unsigned)(((x0 + dx - 1) & 63) * 64 + hi * 16);
        bb1[dx] = (unsigned)((((x0 + 16) + dx - 1) & 63) * 64 + hi * 16);
    }
    const unsigned aoff = (unsigned)lane * 16;  // lane offset inside one A frag

    v8f accs[2][2] = {{v8f{}, v8f{}}, {v8f{}, v8f{}}};

    #pragma unroll
    for (int t = 0; t < 9; ++t) {
        const int ky = t / 3, kx = t % 3;
        const unsigned roff = (unsigned)((py + ky) * 4096);   // halo row (elements)
        #pragma unroll
        for (int kc = 0; kc < 2; ++kc) {
            v16bf bf0 = *(const v16bf*)(xin + roff + bb0[kx] + kc * 32);
            v16bf bf1 = *(const v16bf*)(xin + roff + bb1[kx] + kc * 32);
            #pragma unroll
            for (int mi = 0; mi < 2; ++mi) {
                const int m = mp * 2 + mi;
                v16bf af = *(const v16bf*)(aw + ((unsigned)((m * 9 + t) * 2 + kc) * 512) + aoff);
                accs[mi][0] = __builtin_amdgcn_wmma_f32_16x16x32_bf16(
                    false, af, false, bf0, (short)0, accs[mi][0], false, false);
                accs[mi][1] = __builtin_amdgcn_wmma_f32_16x16x32_bf16(
                    false, af, false, bf1, (short)0, accs[mi][1], false, false);
            }
        }
    }

    // ---- epilogue: FiLM (+ blend + rgb in mode 1) ----
    const float s  = (mode == 1) ? (1.0f / (1.0f + __expf(-blend_p[0]))) : 0.0f;
    const int  yo  = y0 + py;
    #pragma unroll
    for (int mi = 0; mi < 2; ++mi) {
        #pragma unroll
        for (int j = 0; j < 8; ++j) {
            int co = (mp * 2 + mi) * 16 + j + hi * 8;   // C/D layout: M = j (+8 for hi lanes)
            float g  = film[b * 128 + co];
            float be = film[b * 128 + 64 + co];
            #pragma unroll
            for (int ni = 0; ni < 2; ++ni) {
                int xn = x0 + ni * 16;
                float u = fmaxf(fmaf(g, accs[mi][ni][j], be), 0.0f);
                long long oidx = ((long long)(b * CCH + co) * HH + yo) * WWID + xn;
                if (mode == 0) {
                    gout[oidx] = u;
                } else {
                    float go = gold[oidx];
                    float gn = (1.0f - s) * go + s * u;
                    gout[oidx] = gn;
                    if (co < 3) {
                        float rv = 1.0f / (1.0f + __expf(-gn));
                        long long ridx = ((long long)(b * 3 + co) * HH + yo) * WWID + xn;
                        traj[ridx] = rv;
                        if (final_rgb) final_rgb[ridx] = rv;
                    }
                }
            }
        }
    }
}

extern "C" void kernel_launch(void* const* d_in, const int* in_sizes, int n_in,
                              void* d_out, int out_size, void* d_ws, size_t ws_size,
                              hipStream_t stream) {
    const int*   labels = (const int*)  d_in[0];
    const float* initg  = (const float*)d_in[1];
    const float* emb    = (const float*)d_in[2];
    const float* hw1    = (const float*)d_in[3];
    const float* hb1    = (const float*)d_in[4];
    const float* hw2    = (const float*)d_in[5];
    const float* hb2    = (const float*)d_in[6];
    const float* hw3    = (const float*)d_in[7];
    const float* hb3    = (const float*)d_in[8];
    const float* f1w1   = (const float*)d_in[9];
    const float* f1b1   = (const float*)d_in[10];
    const float* f1w2   = (const float*)d_in[11];
    const float* f1b2   = (const float*)d_in[12];
    const float* f2w1   = (const float*)d_in[13];
    const float* f2b1   = (const float*)d_in[14];
    const float* f2w2   = (const float*)d_in[15];
    const float* f2b2   = (const float*)d_in[16];
    const float* blend  = (const float*)d_in[17];

    char* ws = (char*)d_ws;
    float* cond  = (float*)(ws + 0);               // 16 KB
    float* h1    = (float*)(ws + 16384);           // 32 KB
    float* h2    = (float*)(ws + 49152);           // 32 KB
    float* fh    = (float*)(ws + 81920);           // 16 KB (reused per FiLM head)
    float* film1 = (float*)(ws + 98304);           // 16 KB
    float* film2 = (float*)(ws + 114688);          // 16 KB
    float* wraw  = (float*)(ws + 131072);          // 9 MB  [B, 2*TOT1]
    unsigned short* wpack = (unsigned short*)(ws + 9568256);   // 4.5 MB [2][B][TOT1]
    float* gridA = (float*)(ws + 14286848);        // 32 MB
    float* gridB = (float*)(ws + 47841280);        // 32 MB
    float* u1    = (float*)(ws + 81395712);        // 32 MB   (end ~115 MB)

    const size_t GRID_BYTES = (size_t)BATCH * CCH * HH * WWID * sizeof(float);
    hipMemcpyAsync(gridA, initg, GRID_BYTES, hipMemcpyDeviceToDevice, stream);

    // hypernet + FiLM heads (small, plain VALU)
    cond_kernel<<<(BATCH * DDIM + 255) / 256, 256, 0, stream>>>(labels, emb, cond);
    fc_kernel<<<(BATCH * HID) / 256, 256, 0, stream>>>(cond, hw1, hb1, h1, DDIM, HID, 1);
    fc_kernel<<<(BATCH * HID) / 256, 256, 0, stream>>>(h1, hw2, hb2, h2, HID, HID, 1);
    fc_kernel<<<(BATCH * 2 * TOT1) / 256, 256, 0, stream>>>(h2, hw3, hb3, wraw, HID, 2 * TOT1, 0);
    fc_kernel<<<(BATCH * DDIM) / 256, 256, 0, stream>>>(cond, f1w1, f1b1, fh, DDIM, DDIM, 1);
    fc_kernel<<<(BATCH * DDIM) / 256, 256, 0, stream>>>(fh, f1w2, f1b2, film1, DDIM, DDIM, 0);
    fc_kernel<<<(BATCH * DDIM) / 256, 256, 0, stream>>>(cond, f2w1, f2b1, fh, DDIM, DDIM, 1);
    fc_kernel<<<(BATCH * DDIM) / 256, 256, 0, stream>>>(fh, f2w2, f2b2, film2, DDIM, DDIM, 0);
    pack_kernel<<<(2 * BATCH * TOT1) / 256, 256, 0, stream>>>(wraw, wpack);

    const size_t LDS_BYTES = 73728 + 32768;  // A frags + input halo
    float* traj = (float*)d_out + (size_t)BATCH * 3 * HH * WWID;  // after traj[-1] slot
    const size_t RGB_STEP = (size_t)BATCH * 3 * HH * WWID;

    float* gin = gridA;
    float* gout = gridB;
    for (int step = 0; step < NSTEPS; ++step) {
        conv_kernel<<<BATCH * 32, 256, LDS_BYTES, stream>>>(
            gin, nullptr, u1, wpack, film1, blend, nullptr, nullptr, 0);
        float* fin = (step == NSTEPS - 1) ? (float*)d_out : nullptr;
        conv_kernel<<<BATCH * 32, 256, LDS_BYTES, stream>>>(
            u1, gin, gout, wpack + (size_t)BATCH * TOT1, film2, blend,
            traj + (size_t)step * RGB_STEP, fin, 1);
        float* tmp = gin; gin = gout; gout = tmp;
    }
}
